// VectorQuantization_46883863003202
// MI455X (gfx1250) — matrix operand order repaired and checked by reading, not compiled, not run on hardware
//
#include <hip/hip_runtime.h>

typedef _Float16 v16h __attribute__((ext_vector_type(16)));
typedef _Float16 v8h  __attribute__((ext_vector_type(8)));
typedef float    v8f  __attribute__((ext_vector_type(8)));

#define N_PTS 262144
#define DIM   64
#define KCB   512
#define CB_STRIDE 72                       // halfs per codebook row in LDS (padded: 36-dword lane stride, conflict-free)
#define BLOCKS 256
#define THREADS 256
#define ROWS_PER_BLOCK (N_PTS / BLOCKS)    // 1024 rows
#define TILES_PER_WAVE (ROWS_PER_BLOCK / (8 * 16)) // 8 row-tiles of 16

__global__ __launch_bounds__(THREADS) void vq_kernel(const float* __restrict__ x,
                                                     const float* __restrict__ cbg,
                                                     float* __restrict__ out,
                                                     float* __restrict__ partial)
{
    __shared__ _Float16 cb[KCB * CB_STRIDE];   // 73728 B
    __shared__ float    wsq[KCB];              // ||w_k||^2 in fp32
    __shared__ float    sacc[8];

    const int tid = threadIdx.x;

    // ---- Stage 1: codebook f32 -> f16 into LDS, plus exact fp32 row norms ----
    for (int i = tid; i < KCB; i += THREADS) {
        const float* wr = cbg + (size_t)i * DIM;
        float sq = 0.f;
        #pragma unroll
        for (int d = 0; d < DIM; ++d) {
            float v = wr[d];
            sq += v * v;
            cb[i * CB_STRIDE + d] = (_Float16)v;
        }
        wsq[i] = sq;
    }
    __syncthreads();

    const int  wave = tid >> 5;
    const int  lane = tid & 31;
    const int  l15  = lane & 15;
    const bool lolane = (lane < 16);
    float acc = 0.f;

    for (int rt = 0; rt < TILES_PER_WAVE; ++rt) {
        const int rowbase = blockIdx.x * ROWS_PER_BLOCK + wave * (TILES_PER_WAVE * 16) + rt * 16;

        // ---- A fragments (16x32 f16, two k-steps covering D=64) ----
        // ISA layout: lanes 0-15 hold K 0..7 (V0-3) and 16..23 (V4-7); lanes 16-31 hold K 8..15 and 24..31.
        const int rowA = rowbase + l15;
        const int o = lolane ? 0 : 8;
        const float* xr = x + (size_t)rowA * DIM + o;
        v16h a0, a1;
        #pragma unroll
        for (int j = 0; j < 8; ++j) {
            a0[j]     = (_Float16)xr[j];
            a0[j + 8] = (_Float16)xr[j + 16];
            a1[j]     = (_Float16)xr[j + 32];
            a1[j + 8] = (_Float16)xr[j + 48];
        }

        float best[8];
        int   bidx[8];
        #pragma unroll
        for (int r = 0; r < 8; ++r) { best[r] = 3.0e38f; bidx[r] = 0; }

        // ---- sweep 32 codeword tiles of 16 ----
        #pragma unroll 4
        for (int t = 0; t < KCB / 16; ++t) {
            const int col = t * 16 + l15;
            // B (32x16 f16): lane n = column; lanes 0-15 hold K 0..7 and 16..23, lanes 16-31 hold K 8..15 and 24..31.
            const _Float16* bp = &cb[col * CB_STRIDE + (lolane ? 0 : 8)];
            v8h b0l = *(const v8h*)bp;
            v8h b0h = *(const v8h*)(bp + 16);
            v8h b1l = *(const v8h*)(bp + 32);
            v8h b1h = *(const v8h*)(bp + 48);
            v16h b0 = __builtin_shufflevector(b0l, b0h, 0,1,2,3,4,5,6,7,8,9,10,11,12,13,14,15);
            v16h b1 = __builtin_shufflevector(b1l, b1h, 0,1,2,3,4,5,6,7,8,9,10,11,12,13,14,15);
            const float wq = wsq[col];

            v8f c = {0.f,0.f,0.f,0.f,0.f,0.f,0.f,0.f};
            c = __builtin_amdgcn_wmma_f32_16x16x32_f16(false, a0, false, b0, (short)0, c, false, false);
            c = __builtin_amdgcn_wmma_f32_16x16x32_f16(false, a1, false, b1, (short)0, c, false, false);

            // score = ||w||^2 - 2*dot  (||x||^2 constant per row -> irrelevant for argmin)
            #pragma unroll
            for (int r = 0; r < 8; ++r) {
                float s = __builtin_fmaf(-2.f, c[r], wq);
                if (s < best[r]) { best[r] = s; bidx[r] = col; }
            }
        }

        // ---- argmin across the 16 lanes of each half (tie-break: lowest index) ----
        #pragma unroll
        for (int r = 0; r < 8; ++r) {
            float s = best[r]; int id = bidx[r];
            #pragma unroll
            for (int m = 1; m < 16; m <<= 1) {
                float so = __shfl_xor(s, m, 32);
                int   io = __shfl_xor(id, m, 32);
                if (so < s || (so == s && io < id)) { s = so; id = io; }
            }
            best[r] = s; bidx[r] = id;
        }

        // ---- gather (exact fp32), straight-through output, loss ----
        #pragma unroll
        for (int rr = 0; rr < 16; ++rr) {
            const int k   = __shfl(bidx[rr & 7], (rr < 8) ? 0 : 16, 32);
            const int row = rowbase + rr;
            const float2 wv = ((const float2*)(cbg + (size_t)k   * DIM))[lane];
            const float2 xv = ((const float2*)(x   + (size_t)row * DIM))[lane];
            const float t0 = wv.x - xv.x;
            const float t1 = wv.y - xv.y;
            float2 ov; ov.x = xv.x + t0; ov.y = xv.y + t1;   // matches reference rounding
            ((float2*)(out + (size_t)row * DIM))[lane] = ov;
            acc = __builtin_fmaf(t0, t0, acc);
            acc = __builtin_fmaf(t1, t1, acc);
        }
    }

    // ---- deterministic block reduction of squared-error sum ----
    #pragma unroll
    for (int m = 16; m >= 1; m >>= 1) acc += __shfl_xor(acc, m, 32);
    if (lane == 0) sacc[wave] = acc;
    __syncthreads();
    if (tid == 0) {
        float s = 0.f;
        #pragma unroll
        for (int w = 0; w < 8; ++w) s += sacc[w];
        partial[blockIdx.x] = s;
    }
}

__global__ void vq_loss_finalize(const float* __restrict__ partial, float* __restrict__ out)
{
    const int l = threadIdx.x;
    float s = 0.f;
    for (int i = l; i < BLOCKS; i += 32) s += partial[i];
    #pragma unroll
    for (int m = 16; m >= 1; m >>= 1) s += __shfl_xor(s, m, 32);
    if (l == 0)
        out[(size_t)N_PTS * DIM] = 1.25f * (s / ((float)N_PTS * (float)DIM));
}

extern "C" void kernel_launch(void* const* d_in, const int* in_sizes, int n_in,
                              void* d_out, int out_size, void* d_ws, size_t ws_size,
                              hipStream_t stream) {
    const float* x   = (const float*)d_in[0];
    const float* cbg = (const float*)d_in[1];
    float* out       = (float*)d_out;
    float* partial   = (float*)d_ws;   // 256 fp32 partial sums

    vq_kernel<<<BLOCKS, THREADS, 0, stream>>>(x, cbg, out, partial);
    vq_loss_finalize<<<1, 32, 0, stream>>>(partial, out);
}
